// InteractionBlock_77524159693445
// MI455X (gfx1250) — matrix operand order repaired and checked by reading, not compiled, not run on hardware
//
#include <hip/hip_runtime.h>
#include <hip/hip_bf16.h>

#define E_EDGES 262144
#define T_TRIP  2097152
#define HID 128
#define INTC 64

typedef _Float16 v16h __attribute__((ext_vector_type(16)));
typedef float    v8f  __attribute__((ext_vector_type(8)));

#define LSTRIDE 136   // padded LDS row stride in halves (272B = 68 banks offset)

__device__ __forceinline__ float swish_f(float v) {
    return v / (1.0f + __expf(-v));
}

__device__ __forceinline__ v8f wmma_f16(v16h a, v16h b, v8f c) {
    return __builtin_amdgcn_wmma_f32_16x16x32_f16(false, a, false, b, (short)0, c, false, false);
}

// A-fragment (16x32 f16) from an LDS tile laid out [row][col], stride LSTRIDE halves.
// lane<16: elems 0-7 -> K 0-7, elems 8-15 -> K 16-23 (row = lane)
// lane>=16: elems 0-7 -> K 8-15, elems 8-15 -> K 24-31 (row = lane-16)
__device__ __forceinline__ v16h load_a_lds(const _Float16* lds, int lane, int kb) {
    int row = lane & 15;
    int off = (lane < 16) ? 0 : 8;
    const _Float16* p = lds + row * LSTRIDE + kb * 32 + off;
    v16h a;
#pragma unroll
    for (int i = 0; i < 8; ++i) { a[i] = p[i]; a[8 + i] = p[16 + i]; }
    return a;
}

// B-fragment (32x16 f16) from prepped fragment buffer: 32B contiguous per lane.
__device__ __forceinline__ v16h load_b_frag(const _Float16* frag, int lane, int kb, int nb, int NB) {
    return *(const v16h*)(frag + (((size_t)(kb * NB + nb)) * 32 + lane) * 16);
}

// ---------------- weight fragment prep ----------------
// src: KxN row-major f32.  dst layout: [kb][nb][lane][e] f16 where
// col = nb*16 + (lane&15); k = kb*32 + (e<8 ? 8*(lane>>4)+e : 16+8*(lane>>4)+(e-8))
__global__ __launch_bounds__(256) void prep_frag(const float* __restrict__ src,
                                                 _Float16* __restrict__ dst,
                                                 int K, int N) {
    int idx = blockIdx.x * 256 + threadIdx.x;
    if (idx >= K * N) return;
    int e    = idx & 15;
    int lane = (idx >> 4) & 31;
    int rest = idx >> 9;
    int NB = N >> 4;
    int nb = rest % NB;
    int kb = rest / NB;
    int k   = kb * 32 + ((e < 8) ? 8 * (lane >> 4) + e : 16 + 8 * (lane >> 4) + (e - 8));
    int col = nb * 16 + (lane & 15);
    dst[idx] = (_Float16)src[(size_t)k * N + col];
}

__global__ __launch_bounds__(256) void zero_kernel(float* __restrict__ p, int n4) {
    int i = blockIdx.x * 256 + threadIdx.x;
    if (i < n4) {
        float4 z = make_float4(0.f, 0.f, 0.f, 0.f);
        *(float4*)(p + (size_t)i * 4) = z;
    }
}

// ---------------- edge kernel: x_ji, x_down ----------------
__global__ __launch_bounds__(128) void edge_kernel(
    const float* __restrict__ x, const float* __restrict__ rbf,
    const float* __restrict__ W_rbf1, const float* __restrict__ W_rbf2,
    const float* __restrict__ b_ji,   const float* __restrict__ b_kj,
    const _Float16* __restrict__ fW_ji, const _Float16* __restrict__ fW_kj,
    const _Float16* __restrict__ fW_down,
    float* __restrict__ x_ji, float* __restrict__ x_down) {

    __shared__ __attribute__((aligned(16))) float    rbfh[4][16 * 128];
    __shared__ __attribute__((aligned(16))) _Float16 xs[4][16 * LSTRIDE];
    __shared__ __attribute__((aligned(16))) float    t8s[4][16][8];

    const int lane = threadIdx.x & 31;
    const int w    = threadIdx.x >> 5;
    const int base = (blockIdx.x * 4 + w) * 16;
    const int colbase = lane & 15;
    const int rowoff  = (lane < 16) ? 0 : 8;

    // rbf -> t8 (6 -> 8), one row per lane (lanes 0..15)
    if (lane < 16) {
        float t8[8];
#pragma unroll
        for (int b = 0; b < 8; ++b) t8[b] = 0.f;
        const float* rr = rbf + (size_t)(base + lane) * 6;
#pragma unroll
        for (int k = 0; k < 6; ++k) {
            float v = rr[k];
#pragma unroll
            for (int b = 0; b < 8; ++b) t8[b] += v * W_rbf1[k * 8 + b];
        }
#pragma unroll
        for (int b = 0; b < 8; ++b) t8s[w][lane][b] = t8[b];
    }
    __syncthreads();

    // rbf_h tile (16x128) into LDS: t8 @ W_rbf2
    for (int j = 0; j < 4; ++j) {
        int col = j * 32 + lane;
        float wv[8];
#pragma unroll
        for (int b = 0; b < 8; ++b) wv[b] = W_rbf2[b * 128 + col];
        for (int row = 0; row < 16; ++row) {
            float s = 0.f;
#pragma unroll
            for (int b = 0; b < 8; ++b) s += t8s[w][row][b] * wv[b];
            rbfh[w][row * 128 + col] = s;
        }
    }

    // stage x tile (16x128) as f16
    for (int it = 0; it < 16; ++it) {
        int idx4 = it * 32 + lane;          // 0..511 float4s
        int row  = idx4 >> 5;
        int c4   = (idx4 & 31) * 4;
        float4 v = *(const float4*)(x + (size_t)(base + row) * 128 + c4);
        _Float16* d = &xs[w][row * LSTRIDE + c4];
        d[0] = (_Float16)v.x; d[1] = (_Float16)v.y;
        d[2] = (_Float16)v.z; d[3] = (_Float16)v.w;
    }
    __syncthreads();

    // G1 = x@W_ji, G2 = x@W_kj  (shared A loads)
    float xkv[8][8];
    for (int nb = 0; nb < 8; ++nb) {
        v8f acc1 = {}; v8f acc2 = {};
#pragma unroll
        for (int kb = 0; kb < 4; ++kb) {
            v16h a  = load_a_lds(xs[w], lane, kb);
            v16h b1 = load_b_frag(fW_ji, lane, kb, nb, 8);
            v16h b2 = load_b_frag(fW_kj, lane, kb, nb, 8);
            acc1 = wmma_f16(a, b1, acc1);
            acc2 = wmma_f16(a, b2, acc2);
        }
        int col = nb * 16 + colbase;
        float bj = b_ji[col], bk = b_kj[col];
#pragma unroll
        for (int r = 0; r < 8; ++r) {
            int lr = r + rowoff;
            x_ji[(size_t)(base + lr) * 128 + col] = swish_f(acc1[r] + bj);
            xkv[nb][r] = swish_f(acc2[r] + bk) * rbfh[w][lr * 128 + col];
        }
    }
    __syncthreads();

    // restage xk as f16
    for (int nb = 0; nb < 8; ++nb) {
        int col = nb * 16 + colbase;
#pragma unroll
        for (int r = 0; r < 8; ++r)
            xs[w][(r + rowoff) * LSTRIDE + col] = (_Float16)xkv[nb][r];
    }
    __syncthreads();

    // G3 = xk @ W_down (128 -> 64), swish, store x_down
    for (int nb = 0; nb < 4; ++nb) {
        v8f acc = {};
#pragma unroll
        for (int kb = 0; kb < 4; ++kb) {
            v16h a = load_a_lds(xs[w], lane, kb);
            v16h b = load_b_frag(fW_down, lane, kb, nb, 4);
            acc = wmma_f16(a, b, acc);
        }
        int col = nb * 16 + colbase;
#pragma unroll
        for (int r = 0; r < 8; ++r)
            x_down[(size_t)(base + r + rowoff) * 64 + col] = swish_f(acc[r]);
    }
}

// ---------------- triplet kernel: gather * sbf_h -> scatter-add ----------------
__global__ __launch_bounds__(256) void triplet_kernel(
    const float* __restrict__ sbf, const int* __restrict__ idx_kj,
    const int* __restrict__ idx_ji,
    const float* __restrict__ W_sbf1, const float* __restrict__ W_sbf2,
    const float* __restrict__ x_down, float* __restrict__ agg) {

    __shared__ float w1[42 * 8];
    __shared__ float w2[8 * 64];
    for (int i = threadIdx.x; i < 336; i += 256) w1[i] = W_sbf1[i];
    for (int i = threadIdx.x; i < 512; i += 256) w2[i] = W_sbf2[i];
    __syncthreads();

    int t = blockIdx.x * 256 + threadIdx.x;
    int kj = idx_kj[t];
    int ji = idx_ji[t];
    const float* srow = sbf + (size_t)t * 42;

    float t8[8];
#pragma unroll
    for (int b = 0; b < 8; ++b) t8[b] = 0.f;
    for (int k = 0; k < 42; ++k) {
        float v = srow[k];
#pragma unroll
        for (int b = 0; b < 8; ++b) t8[b] += v * w1[k * 8 + b];
    }

    const float* xd = x_down + (size_t)kj * 64;
    float* ag = agg + (size_t)ji * 64;
    for (int c = 0; c < 64; c += 4) {
        float4 xv = *(const float4*)(xd + c);
        float s0 = 0.f, s1 = 0.f, s2 = 0.f, s3 = 0.f;
#pragma unroll
        for (int b = 0; b < 8; ++b) {
            const float* wp = &w2[b * 64 + c];
            s0 += t8[b] * wp[0]; s1 += t8[b] * wp[1];
            s2 += t8[b] * wp[2]; s3 += t8[b] * wp[3];
        }
        unsafeAtomicAdd(ag + c + 0, xv.x * s0);
        unsafeAtomicAdd(ag + c + 1, xv.y * s1);
        unsafeAtomicAdd(ag + c + 2, xv.z * s2);
        unsafeAtomicAdd(ag + c + 3, xv.w * s3);
    }
}

// ---------------- output chain helpers ----------------
__device__ __forceinline__ void stage_tile(float (&h)[8][8], _Float16* hsw, int lane) {
    __syncthreads();
    int colbase = lane & 15, rowoff = (lane < 16) ? 0 : 8;
#pragma unroll
    for (int nb = 0; nb < 8; ++nb)
#pragma unroll
        for (int r = 0; r < 8; ++r)
            hsw[(r + rowoff) * LSTRIDE + nb * 16 + colbase] = (_Float16)h[nb][r];
    __syncthreads();
}

__device__ __forceinline__ void residual_step(float (&h)[8][8], _Float16* hsw,
    const _Float16* fW1, const float* b1, const _Float16* fW2, const float* b2, int lane) {
    int colbase = lane & 15;
    float t[8][8];
    for (int nb = 0; nb < 8; ++nb) {
        v8f acc = {};
#pragma unroll
        for (int kb = 0; kb < 4; ++kb) {
            v16h a = load_a_lds(hsw, lane, kb);
            v16h b = load_b_frag(fW1, lane, kb, nb, 8);
            acc = wmma_f16(a, b, acc);
        }
        float bb = b1[nb * 16 + colbase];
#pragma unroll
        for (int r = 0; r < 8; ++r) t[nb][r] = swish_f(acc[r] + bb);
    }
    __syncthreads();
    int rowoff = (lane < 16) ? 0 : 8;
#pragma unroll
    for (int nb = 0; nb < 8; ++nb)
#pragma unroll
        for (int r = 0; r < 8; ++r)
            hsw[(r + rowoff) * LSTRIDE + nb * 16 + colbase] = (_Float16)t[nb][r];
    __syncthreads();
    for (int nb = 0; nb < 8; ++nb) {
        v8f acc = {};
#pragma unroll
        for (int kb = 0; kb < 4; ++kb) {
            v16h a = load_a_lds(hsw, lane, kb);
            v16h b = load_b_frag(fW2, lane, kb, nb, 8);
            acc = wmma_f16(a, b, acc);
        }
        float bb = b2[nb * 16 + colbase];
#pragma unroll
        for (int r = 0; r < 8; ++r) h[nb][r] += swish_f(acc[r] + bb);
    }
}

// ---------------- output kernel: up + residual chain ----------------
__global__ __launch_bounds__(128) void out_kernel(
    const float* __restrict__ x, const float* __restrict__ agg,
    const float* __restrict__ x_ji,
    const _Float16* __restrict__ fW_up, const _Float16* __restrict__ fW_lin,
    const _Float16* __restrict__ frb0W1, const _Float16* __restrict__ frb0W2,
    const _Float16* __restrict__ fra0W1, const _Float16* __restrict__ fra0W2,
    const _Float16* __restrict__ fra1W1, const _Float16* __restrict__ fra1W2,
    const float* __restrict__ b_lin,
    const float* __restrict__ rb0b1, const float* __restrict__ rb0b2,
    const float* __restrict__ ra0b1, const float* __restrict__ ra0b2,
    const float* __restrict__ ra1b1, const float* __restrict__ ra1b2,
    float* __restrict__ out) {

    __shared__ __attribute__((aligned(16))) _Float16 hs[4][16 * LSTRIDE];

    const int lane = threadIdx.x & 31;
    const int w    = threadIdx.x >> 5;
    const int base = (blockIdx.x * 4 + w) * 16;
    const int colbase = lane & 15;
    const int rowoff  = (lane < 16) ? 0 : 8;

    // stage agg tile (16x64) as f16
    for (int it = 0; it < 8; ++it) {
        int idx4 = it * 32 + lane;          // 0..255 float4s
        int row  = idx4 >> 4;
        int c4   = (idx4 & 15) * 4;
        float4 v = *(const float4*)(agg + (size_t)(base + row) * 64 + c4);
        _Float16* d = &hs[w][row * LSTRIDE + c4];
        d[0] = (_Float16)v.x; d[1] = (_Float16)v.y;
        d[2] = (_Float16)v.z; d[3] = (_Float16)v.w;
    }
    __syncthreads();

    // h = x_ji + swish(agg @ W_up)   (K = 64)
    float h[8][8];
    for (int nb = 0; nb < 8; ++nb) {
        v8f acc = {};
#pragma unroll
        for (int kb = 0; kb < 2; ++kb) {
            v16h a = load_a_lds(hs[w], lane, kb);
            v16h b = load_b_frag(fW_up, lane, kb, nb, 8);
            acc = wmma_f16(a, b, acc);
        }
        int col = nb * 16 + colbase;
#pragma unroll
        for (int r = 0; r < 8; ++r)
            h[nb][r] = x_ji[(size_t)(base + r + rowoff) * 128 + col] + swish_f(acc[r]);
    }
    stage_tile(h, hs[w], lane);

    // res_before
    residual_step(h, hs[w], frb0W1, rb0b1, frb0W2, rb0b2, lane);
    stage_tile(h, hs[w], lane);

    // h = swish(h @ W_lin + b_lin) + x
    {
        float t[8][8];
        for (int nb = 0; nb < 8; ++nb) {
            v8f acc = {};
#pragma unroll
            for (int kb = 0; kb < 4; ++kb) {
                v16h a = load_a_lds(hs[w], lane, kb);
                v16h b = load_b_frag(fW_lin, lane, kb, nb, 8);
                acc = wmma_f16(a, b, acc);
            }
            int col = nb * 16 + colbase;
            float bb = b_lin[col];
#pragma unroll
            for (int r = 0; r < 8; ++r)
                t[nb][r] = swish_f(acc[r] + bb) + x[(size_t)(base + r + rowoff) * 128 + col];
        }
#pragma unroll
        for (int nb = 0; nb < 8; ++nb)
#pragma unroll
            for (int r = 0; r < 8; ++r) h[nb][r] = t[nb][r];
    }
    stage_tile(h, hs[w], lane);

    // res_after x2
    residual_step(h, hs[w], fra0W1, ra0b1, fra0W2, ra0b2, lane);
    stage_tile(h, hs[w], lane);
    residual_step(h, hs[w], fra1W1, ra1b1, fra1W2, ra1b2, lane);

    // store
    for (int nb = 0; nb < 8; ++nb) {
        int col = nb * 16 + colbase;
#pragma unroll
        for (int r = 0; r < 8; ++r)
            out[(size_t)(base + r + rowoff) * 128 + col] = h[nb][r];
    }
}

extern "C" void kernel_launch(void* const* d_in, const int* in_sizes, int n_in,
                              void* d_out, int out_size, void* d_ws, size_t ws_size,
                              hipStream_t stream) {
    const float* x      = (const float*)d_in[0];
    const float* rbf    = (const float*)d_in[1];
    const float* sbf    = (const float*)d_in[2];
    const int*   idx_kj = (const int*)d_in[3];
    const int*   idx_ji = (const int*)d_in[4];
    const float* W_rbf1 = (const float*)d_in[5];
    const float* W_rbf2 = (const float*)d_in[6];
    const float* W_sbf1 = (const float*)d_in[7];
    const float* W_sbf2 = (const float*)d_in[8];
    const float* W_kj   = (const float*)d_in[9];
    const float* b_kj   = (const float*)d_in[10];
    const float* W_ji   = (const float*)d_in[11];
    const float* b_ji   = (const float*)d_in[12];
    const float* W_down = (const float*)d_in[13];
    const float* W_up   = (const float*)d_in[14];
    const float* W_lin  = (const float*)d_in[15];
    const float* b_lin  = (const float*)d_in[16];
    const float* rb0W1  = (const float*)d_in[17];
    const float* rb0b1  = (const float*)d_in[18];
    const float* rb0W2  = (const float*)d_in[19];
    const float* rb0b2  = (const float*)d_in[20];
    const float* ra0W1  = (const float*)d_in[21];
    const float* ra0b1  = (const float*)d_in[22];
    const float* ra0W2  = (const float*)d_in[23];
    const float* ra0b2  = (const float*)d_in[24];
    const float* ra1W1  = (const float*)d_in[25];
    const float* ra1b1  = (const float*)d_in[26];
    const float* ra1W2  = (const float*)d_in[27];
    const float* ra1b2  = (const float*)d_in[28];

    // workspace layout
    _Float16* fb = (_Float16*)d_ws;
    _Float16* f_ji    = fb + 0;
    _Float16* f_kj    = fb + 16384;
    _Float16* f_down  = fb + 32768;
    _Float16* f_up    = fb + 40960;
    _Float16* f_lin   = fb + 49152;
    _Float16* f_rb0w1 = fb + 65536;
    _Float16* f_rb0w2 = fb + 81920;
    _Float16* f_ra0w1 = fb + 98304;
    _Float16* f_ra0w2 = fb + 114688;
    _Float16* f_ra1w1 = fb + 131072;
    _Float16* f_ra1w2 = fb + 147456;          // end: 163840 halves = 327680 B
    float* x_ji_buf   = (float*)((char*)d_ws + 327680);
    float* x_down_buf = x_ji_buf + (size_t)E_EDGES * 128;
    float* agg_buf    = x_down_buf + (size_t)E_EDGES * 64;

    // weight fragment prep
    prep_frag<<<64, 256, 0, stream>>>(W_ji,   f_ji,    128, 128);
    prep_frag<<<64, 256, 0, stream>>>(W_kj,   f_kj,    128, 128);
    prep_frag<<<32, 256, 0, stream>>>(W_down, f_down,  128, 64);
    prep_frag<<<32, 256, 0, stream>>>(W_up,   f_up,    64,  128);
    prep_frag<<<64, 256, 0, stream>>>(W_lin,  f_lin,   128, 128);
    prep_frag<<<64, 256, 0, stream>>>(rb0W1,  f_rb0w1, 128, 128);
    prep_frag<<<64, 256, 0, stream>>>(rb0W2,  f_rb0w2, 128, 128);
    prep_frag<<<64, 256, 0, stream>>>(ra0W1,  f_ra0w1, 128, 128);
    prep_frag<<<64, 256, 0, stream>>>(ra0W2,  f_ra0w2, 128, 128);
    prep_frag<<<64, 256, 0, stream>>>(ra1W1,  f_ra1w1, 128, 128);
    prep_frag<<<64, 256, 0, stream>>>(ra1W2,  f_ra1w2, 128, 128);

    // zero the aggregation buffer (E*64 floats = 4194304 float4s)
    zero_kernel<<<(E_EDGES * 64 / 4 + 255) / 256, 256, 0, stream>>>(agg_buf, E_EDGES * 64 / 4);

    // edge GEMMs
    edge_kernel<<<E_EDGES / 64, 128, 0, stream>>>(
        x, rbf, W_rbf1, W_rbf2, b_ji, b_kj, f_ji, f_kj, f_down, x_ji_buf, x_down_buf);

    // triplet aggregation
    triplet_kernel<<<T_TRIP / 256, 256, 0, stream>>>(
        sbf, idx_kj, idx_ji, W_sbf1, W_sbf2, x_down_buf, agg_buf);

    // output chain
    out_kernel<<<E_EDGES / 64, 128, 0, stream>>>(
        x, agg_buf, x_ji_buf, f_up, f_lin,
        f_rb0w1, f_rb0w2, f_ra0w1, f_ra0w2, f_ra1w1, f_ra1w2,
        b_lin, rb0b1, rb0b2, ra0b1, ra0b2, ra1b1, ra1b2,
        (float*)d_out);
}
